// StandardAttention_6571299963365
// MI455X (gfx1250) — matrix operand order repaired and checked by reading, not compiled, not run on hardware
//
#include <hip/hip_runtime.h>

// Flash attention for MI455X (gfx1250, wave32, WMMA f16->f32), transposed
// formulation: S^T = K*Q^T, O^T = V^T*P^T, so softmax reduction over keys is
// in-lane (per-query lane-scalar state), with only 2 cross-lane shuffles/tile.

typedef __attribute__((ext_vector_type(16))) _Float16 v16h;
typedef __attribute__((ext_vector_type(8)))  _Float16 v8h;
typedef __attribute__((ext_vector_type(4)))  _Float16 v4h;
typedef __attribute__((ext_vector_type(8)))  float    v8f;
typedef __attribute__((ext_vector_type(4)))  float    f32x4;

#define SEQ     4096
#define DMODEL  64
#define KT      32                 // keys per iteration
#define MW      16                 // queries per wave (WMMA N)
#define NWAVES  4
#define MB      (MW * NWAVES)
#define NT      (SEQ / KT)
// 1/sqrt(64) * log2(e): exp2-domain softmax, folded into Q at load.
#define QSCALE  0.180336880f

__global__ __launch_bounds__(128)
void fa_fwd_kernel(const float* __restrict__ Q,
                   const float* __restrict__ K,
                   const float* __restrict__ V,
                   float* __restrict__ O) {
    // Double-buffered K (row-major [key][d]) and V^T ([d][key]) tiles, f16.
    __shared__ __align__(64) _Float16 Ks[2][KT * DMODEL];      // 2 x 4 KB
    __shared__ __align__(64) _Float16 Vt[2][DMODEL * KT];      // 2 x 4 KB
    __shared__ __align__(64) _Float16 Pb[NWAVES * MW * KT];    // 4 KB (per-wave)

    const int tid  = threadIdx.x;
    const int wave = tid >> 5;
    const int lane = tid & 31;
    const int lh   = lane >> 4;    // 16-lane half
    const int ll   = lane & 15;    // query within wave tile / column index

    const int b    = blockIdx.y;
    const int q0w  = blockIdx.x * MB + wave * MW;   // wave's query base

    const float* Qb = Q + (size_t)b * SEQ * DMODEL;
    const float* Kb = K + (size_t)b * SEQ * DMODEL;
    const float* Vb = V + (size_t)b * SEQ * DMODEL;
    float*       Ob = O + (size_t)b * SEQ * DMODEL;

    _Float16* pw = Pb + wave * (MW * KT);

    // ---- Q as B-matrix fragments (loop invariant, exp2+scale folded) ----
    // B 32x16: lane col N=ll (query), lanes 0-15 K(d)=0..15, lanes 16-31 K=16..31.
    v16h bq[2];
    {
        const float* qp = Qb + (size_t)(q0w + ll) * DMODEL;
        #pragma unroll
        for (int c = 0; c < 2; ++c) {
            const float* qc = qp + 32 * c + 16 * lh;
            #pragma unroll
            for (int j = 0; j < 4; ++j) {
                f32x4 qv = *(const f32x4*)(qc + 4 * j);
                #pragma unroll
                for (int e = 0; e < 4; ++e)
                    bq[c][4 * j + e] = (_Float16)(qv[e] * QSCALE);
            }
        }
    }

    // Cooperative-load index schemes.
    const int krow = tid >> 2;            // K: 0..31 key row
    const int kcb  = (tid & 3) * 16;      // K: d column base (16 floats)
    const int vk0  = (tid >> 4) * 4;      // V: 4-key patch base
    const int vd0  = (tid & 15) * 4;      // V: 4-d patch base

    f32x4 kr[4], vr[4];
    auto load_tile = [&](int it) {
        const float* kp = Kb + (size_t)(it * KT + krow) * DMODEL + kcb;
        #pragma unroll
        for (int j = 0; j < 4; ++j) kr[j] = *(const f32x4*)(kp + 4 * j);
        const float* vp = Vb + (size_t)(it * KT + vk0) * DMODEL + vd0;
        #pragma unroll
        for (int i = 0; i < 4; ++i)
            vr[i] = *(const f32x4*)(vp + (size_t)i * DMODEL);
    };
    auto store_tile = [&](int buf) {
        // K row-major f16: 16 contiguous halves = 2 x b128.
        _Float16* kd = &Ks[buf][krow * DMODEL + kcb];
        v8h h0, h1;
        #pragma unroll
        for (int e = 0; e < 4; ++e) {
            h0[e] = (_Float16)kr[0][e]; h0[4 + e] = (_Float16)kr[1][e];
            h1[e] = (_Float16)kr[2][e]; h1[4 + e] = (_Float16)kr[3][e];
        }
        *(v8h*)(kd)     = h0;
        *(v8h*)(kd + 8) = h1;
        // V transposed f16: pack 4 keys per d -> 4 x b64.
        #pragma unroll
        for (int i = 0; i < 4; ++i) {
            v4h t;
            #pragma unroll
            for (int e = 0; e < 4; ++e) t[e] = (_Float16)vr[e][i];
            *(v4h*)&Vt[buf][(vd0 + i) * KT + vk0] = t;
        }
    };

    // Online-softmax lane-scalar state (per query ll) + O^T accumulators:
    // o[t] VGPR r, lane l -> d = 16t + r + 8*lh, query = ll.
    float m_i = -3.0e38f, l_i = 0.0f;
    v8f o[4] = {};

    // Prologue: stage tile 0.
    load_tile(0);
    store_tile(0);
    __syncthreads();

    for (int it = 0; it < NT; ++it) {
        const int cur = it & 1;
        const bool more = (it + 1) < NT;
        if (more) load_tile(it + 1);           // hide global latency

        // ---- S^T = K Q^T : two 16-key subtiles, 2 x K32 d-chunks each ----
        const _Float16* ks = &Ks[cur][0];
        v8f st[2];
        #pragma unroll
        for (int sb = 0; sb < 2; ++sb) {
            v8f acc = {};
            #pragma unroll
            for (int c = 0; c < 2; ++c) {
                // A 16x32: lane row M=key; d = 32c+16lh..+15 contiguous.
                const _Float16* ap = ks + (sb * 16 + ll) * DMODEL + 32 * c + 16 * lh;
                v8h lo = *(const v8h*)(ap);
                v8h hi = *(const v8h*)(ap + 8);
                v16h ak;
                #pragma unroll
                for (int i = 0; i < 8; ++i) { ak[i] = lo[i]; ak[8 + i] = hi[i]; }
                acc = __builtin_amdgcn_wmma_f32_16x16x32_f16(
                        false, ak, false, bq[c], (short)0, acc, false, false);
            }
            st[sb] = acc;
        }

        // ---- Online softmax (exp2 domain), in-lane over 16 keys ----
        float tm = fmaxf(st[0][0], st[1][0]);
        #pragma unroll
        for (int r = 1; r < 8; ++r)
            tm = fmaxf(tm, fmaxf(st[0][r], st[1][r]));
        tm = fmaxf(tm, __shfl_xor(tm, 16, 32));        // other half's keys
        const float mnew = fmaxf(m_i, tm);
        const float corr = __builtin_amdgcn_exp2f(m_i - mnew);
        m_i = mnew;

        v8h ph0, ph1;
        float rs = 0.0f;
        #pragma unroll
        for (int r = 0; r < 8; ++r) {
            const float p0 = __builtin_amdgcn_exp2f(st[0][r] - mnew);
            const float p1 = __builtin_amdgcn_exp2f(st[1][r] - mnew);
            ph0[r] = (_Float16)p0;
            ph1[r] = (_Float16)p1;
            rs += p0 + p1;
        }
        rs += __shfl_xor(rs, 16, 32);
        l_i = l_i * corr + rs;
        #pragma unroll
        for (int t = 0; t < 4; ++t) o[t] *= corr;

        // ---- P^T half-exchange through per-wave LDS (keys contiguous) ----
        // Row = query ll (32 keys); this lane owns keys r+8lh and 16+r+8lh.
        // LDS ops are in-order within a wave and these provably alias the
        // B-fragment load below, so program order alone guarantees the RAW;
        // the compiler inserts the minimal s_wait_dscnt before use.
        *(v8h*)(pw + ll * KT + 8 * lh)      = ph0;
        *(v8h*)(pw + ll * KT + 16 + 8 * lh) = ph1;
        // B 32x16: lane col = query ll; K(keys) = 16lh..16lh+15 contiguous.
        v16h pb = *(const v16h*)(pw + ll * KT + 16 * lh);

        // ---- O^T += V^T P^T : 4 d-subtiles, K = 32 keys ----
        const _Float16* vt = &Vt[cur][0];
        #pragma unroll
        for (int t = 0; t < 4; ++t) {
            // A 16x32: lane row M = d = 16t+ll; keys 16lh..16lh+15 contiguous.
            const _Float16* ap = vt + (16 * t + ll) * KT + 16 * lh;
            v8h lo = *(const v8h*)(ap);
            v8h hi = *(const v8h*)(ap + 8);
            v16h av;
            #pragma unroll
            for (int i = 0; i < 8; ++i) { av[i] = lo[i]; av[8 + i] = hi[i]; }
            o[t] = __builtin_amdgcn_wmma_f32_16x16x32_f16(
                     false, av, false, pb, (short)0, o[t], false, false);
        }

        // ---- Stage next tile into the other buffer; one barrier/iter ----
        if (more) store_tile(cur ^ 1);
        __syncthreads();
    }

    // ---- Normalize and store O (row = query ll, contiguous d per lane) ----
    const float inv = 1.0f / l_i;
    float* op = Ob + (size_t)(q0w + ll) * DMODEL + 8 * lh;
    #pragma unroll
    for (int t = 0; t < 4; ++t) {
        f32x4 w0, w1;
        #pragma unroll
        for (int e = 0; e < 4; ++e) {
            w0[e] = o[t][e] * inv;
            w1[e] = o[t][4 + e] * inv;
        }
        *(f32x4*)(op + 16 * t)     = w0;
        *(f32x4*)(op + 16 * t + 4) = w1;
    }
}

extern "C" void kernel_launch(void* const* d_in, const int* in_sizes, int n_in,
                              void* d_out, int out_size, void* d_ws, size_t ws_size,
                              hipStream_t stream) {
    (void)in_sizes; (void)n_in; (void)out_size; (void)d_ws; (void)ws_size;
    const float* q = (const float*)d_in[0];
    const float* k = (const float*)d_in[1];
    const float* v = (const float*)d_in[2];
    float* o = (float*)d_out;

    dim3 grid(SEQ / MB, 4, 1);   // 64 query-blocks x 4 batches
    dim3 block(128, 1, 1);       // 4 waves (wave32), 16 queries each
    hipLaunchKernelGGL(fa_fwd_kernel, grid, block, 0, stream, q, k, v, o);
}
